// GCNRegression_29437705847600
// MI455X (gfx1250) — matrix operand order repaired and checked by reading, not compiled, not run on hardware
//
#include <hip/hip_runtime.h>
#include <hip/hip_bf16.h>

typedef __attribute__((ext_vector_type(16))) _Float16 v16h;
typedef __attribute__((ext_vector_type(8)))  float    v8f;

#define DFEAT 64

// ---------------- degree / norm precompute ----------------

__global__ void k_init_deg(float* __restrict__ deg, int n) {
    int i = blockIdx.x * blockDim.x + threadIdx.x;
    if (i < n) deg[i] = 1.0f;  // self-loop weight
}

__global__ void k_acc_deg(const int* __restrict__ dst, const float* __restrict__ w,
                          float* __restrict__ deg, int E) {
    int e = blockIdx.x * blockDim.x + threadIdx.x;
    if (e < E) atomicAdd(&deg[dst[e]], w[e]);
}

__global__ void k_dinv(const float* __restrict__ deg, float* __restrict__ dinv, int n) {
    int i = blockIdx.x * blockDim.x + threadIdx.x;
    if (i < n) {
        float d = deg[i];
        dinv[i] = (d > 0.0f) ? rsqrtf(d) : 0.0f;
    }
}

__global__ void k_norm(const int* __restrict__ src, const int* __restrict__ dst,
                       const float* __restrict__ w, const float* __restrict__ dinv,
                       float* __restrict__ nrm, int E) {
    int e = blockIdx.x * blockDim.x + threadIdx.x;
    if (e < E) nrm[e] = dinv[src[e]] * w[e] * dinv[dst[e]];
}

// ---------------- 64x64 GEMM via WMMA (f16 in, f32 acc) ----------------
// One wave computes a 16x16 tile of H = X @ W.  4 waves/block cover all 64
// output columns for 16 rows.  K = 64 -> two v_wmma_f32_16x16x32_f16.

__global__ void k_gemm64_wmma(const float* __restrict__ X, const float* __restrict__ W,
                              float* __restrict__ H, int n) {
    const int wave  = threadIdx.x >> 5;
    const int lane  = threadIdx.x & 31;
    const int l16   = lane & 15;     // A-row within tile / B,D column within tile
    const int khalf = lane >> 4;     // which K-half this lane holds
    const int m0 = blockIdx.x * 16;
    if (m0 >= n) return;             // uniform per block -> EXEC stays all-1s
    const int n0 = wave * 16;

    v8f acc = {};
#pragma unroll
    for (int kb = 0; kb < 2; ++kb) {
        // A fragment: 16x32 f16.  Lane holds row m0+l16.
        // elems 0..7  -> K = kb*32 + khalf*8 + j        (8 contiguous floats)
        // elems 8..15 -> K = kb*32 + 16 + khalf*8 + j   (8 contiguous floats)
        v16h a;
        const float* xr = X + (size_t)(m0 + l16) * DFEAT + kb * 32 + khalf * 8;
#pragma unroll
        for (int j = 0; j < 8; ++j) a[j] = (_Float16)xr[j];
#pragma unroll
        for (int j = 0; j < 8; ++j) a[8 + j] = (_Float16)xr[16 + j];

        // B fragment: 32x16 f16.  Lane holds column n0+l16.
        // elem j -> K = kb*32 + khalf*8 + j (j<8), kb*32 + 16 + khalf*8 + (j-8) (j>=8)   (stride-64 column walk)
        v16h b;
        const float* wc = W + (size_t)(kb * 32 + khalf * 8) * DFEAT + n0 + l16;
#pragma unroll
        for (int j = 0; j < 16; ++j) b[j] = (_Float16)wc[(size_t)(j + (j & 8)) * DFEAT];

        acc = __builtin_amdgcn_wmma_f32_16x16x32_f16(
            /*neg_a=*/false, a, /*neg_b=*/false, b,
            /*c_mod=*/(short)0, acc, /*reuse_a=*/false, /*reuse_b=*/false);
    }

    // D layout: vgpr v holds row M = v + 8*khalf, column N = l16.
#pragma unroll
    for (int v = 0; v < 8; ++v)
        H[(size_t)(m0 + v + 8 * khalf) * DFEAT + n0 + l16] = acc[v];
}

// ---------------- aggregation ----------------

// OUT[i] = dinv[i]^2 * H[i]   (self-loop contribution doubles as initializer)
__global__ void k_selfloop_init(const float* __restrict__ H, const float* __restrict__ dinv,
                                float* __restrict__ OUT, int n) {
    int t = blockIdx.x * blockDim.x + threadIdx.x;   // n*16 threads, float4 each
    int i = t >> 4;
    if (i >= n) return;
    int c = (t & 15) << 2;
    float s = dinv[i]; s = s * s;
    float4 h = *(const float4*)(H + (size_t)i * DFEAT + c);
    float4 o = { s * h.x, s * h.y, s * h.z, s * h.w };
    *(float4*)(OUT + (size_t)i * DFEAT + c) = o;
}

// OUT[dst] += norm * H[src], 16 lanes per edge, float4 per lane
__global__ void k_edge_scatter(const int* __restrict__ src, const int* __restrict__ dst,
                               const float* __restrict__ nrm, const float* __restrict__ H,
                               float* __restrict__ OUT, int E) {
    int t = blockIdx.x * blockDim.x + threadIdx.x;
    int e = t >> 4;
    if (e >= E) return;
    int c = (t & 15) << 2;
    int s = src[e], d = dst[e];
    float w = nrm[e];
    float4 h = *(const float4*)(H + (size_t)s * DFEAT + c);
    float* o = OUT + (size_t)d * DFEAT + c;
    atomicAdd(o + 0, w * h.x);
    atomicAdd(o + 1, w * h.y);
    atomicAdd(o + 2, w * h.z);
    atomicAdd(o + 3, w * h.w);
}

__global__ void k_bias_relu(float* __restrict__ OUT, const float* __restrict__ b, int total) {
    int t = blockIdx.x * blockDim.x + threadIdx.x;
    if (t >= total) return;
    float v = OUT[t] + b[t & (DFEAT - 1)];
    OUT[t] = v > 0.0f ? v : 0.0f;
}

// ---------------- layer 3: GEMV + scalar aggregation ----------------

__global__ void k_gemv(const float* __restrict__ H, const float* __restrict__ W3,
                       float* __restrict__ s, int n) {
    int i = blockIdx.x * blockDim.x + threadIdx.x;
    if (i >= n) return;
    const float4* h4 = (const float4*)(H + (size_t)i * DFEAT);
    const float4* w4 = (const float4*)W3;
    float acc = 0.0f;
#pragma unroll
    for (int j = 0; j < DFEAT / 4; ++j) {
        float4 hv = h4[j], wv = w4[j];
        acc += hv.x * wv.x + hv.y * wv.y + hv.z * wv.z + hv.w * wv.w;
    }
    s[i] = acc;
}

__global__ void k_selfloop_init_scalar(const float* __restrict__ s, const float* __restrict__ dinv,
                                       float* __restrict__ OUT, int n) {
    int i = blockIdx.x * blockDim.x + threadIdx.x;
    if (i >= n) return;
    float d = dinv[i];
    OUT[i] = d * d * s[i];
}

__global__ void k_edge_scatter_scalar(const int* __restrict__ src, const int* __restrict__ dst,
                                      const float* __restrict__ nrm, const float* __restrict__ s,
                                      float* __restrict__ OUT, int E) {
    int e = blockIdx.x * blockDim.x + threadIdx.x;
    if (e >= E) return;
    atomicAdd(&OUT[dst[e]], nrm[e] * s[src[e]]);
}

__global__ void k_sigmoid(float* __restrict__ OUT, const float* __restrict__ b3, int n) {
    int i = blockIdx.x * blockDim.x + threadIdx.x;
    if (i >= n) return;
    float v = OUT[i] + b3[0];
    OUT[i] = 1.0f / (1.0f + __expf(-v));
}

// ---------------- driver ----------------

extern "C" void kernel_launch(void* const* d_in, const int* in_sizes, int n_in,
                              void* d_out, int out_size, void* d_ws, size_t ws_size,
                              hipStream_t stream) {
    const float* x   = (const float*)d_in[0];
    const int*   ei  = (const int*)d_in[1];
    const float* ew  = (const float*)d_in[2];
    const float* W1  = (const float*)d_in[3];
    const float* b1  = (const float*)d_in[4];
    const float* W2  = (const float*)d_in[5];
    const float* b2  = (const float*)d_in[6];
    const float* W3  = (const float*)d_in[7];
    const float* b3  = (const float*)d_in[8];

    const int N = in_sizes[0] / DFEAT;
    const int E = in_sizes[2];
    const int* src = ei;
    const int* dst = ei + E;

    float* ws   = (float*)d_ws;
    float* deg  = ws;                 ws += N;
    float* dinv = ws;                 ws += N;
    float* nrm  = ws;                 ws += E;
    float* hA   = ws;                 ws += (size_t)N * DFEAT;
    float* hB   = ws;                 ws += (size_t)N * DFEAT;
    float* sA   = ws;                 ws += N;
    float* out  = (float*)d_out;

    const int B = 256;
    const int gN   = (N + B - 1) / B;
    const int gE   = (E + B - 1) / B;
    const int gN16 = (N * 16 + B - 1) / B;
    const int gE16 = (E * 16 + B - 1) / B;
    const int gND  = (N * DFEAT + B - 1) / B;
    const int gGemm = (N + 15) / 16;      // 16 rows per block, 4 waves cover 64 cols

    // normalization (once, reused for all 3 layers)
    k_init_deg<<<gN, B, 0, stream>>>(deg, N);
    k_acc_deg<<<gE, B, 0, stream>>>(dst, ew, deg, E);
    k_dinv<<<gN, B, 0, stream>>>(deg, dinv, N);
    k_norm<<<gE, B, 0, stream>>>(src, dst, ew, dinv, nrm, E);

    // layer 1: hA = x @ W1 ; hB = aggregate(hA) ; relu(hB + b1)
    k_gemm64_wmma<<<gGemm, 128, 0, stream>>>(x, W1, hA, N);
    k_selfloop_init<<<gN16, B, 0, stream>>>(hA, dinv, hB, N);
    k_edge_scatter<<<gE16, B, 0, stream>>>(src, dst, nrm, hA, hB, E);
    k_bias_relu<<<gND, B, 0, stream>>>(hB, b1, N * DFEAT);

    // layer 2: hA = hB @ W2 ; hB = aggregate(hA) ; relu(hB + b2)
    k_gemm64_wmma<<<gGemm, 128, 0, stream>>>(hB, W2, hA, N);
    k_selfloop_init<<<gN16, B, 0, stream>>>(hA, dinv, hB, N);
    k_edge_scatter<<<gE16, B, 0, stream>>>(src, dst, nrm, hA, hB, E);
    k_bias_relu<<<gND, B, 0, stream>>>(hB, b2, N * DFEAT);

    // layer 3: sA = hB @ W3 ; out = aggregate(sA) ; sigmoid(out + b3)
    k_gemv<<<gN, B, 0, stream>>>(hB, W3, sA, N);
    k_selfloop_init_scalar<<<gN, B, 0, stream>>>(sA, dinv, out, N);
    k_edge_scatter_scalar<<<gE, B, 0, stream>>>(src, dst, nrm, sA, out, E);
    k_sigmoid<<<gN, B, 0, stream>>>(out, b3, N);
}
